// MyModel_15599321219271
// MI455X (gfx1250) — compile-verified, matrix operand order checked by reading, not run on hardware
//
#include <hip/hip_runtime.h>
#include <hip/hip_bf16.h>

// ---------------- problem constants ----------------
#define BB   256      // batch
#define TT   256      // time
#define EE   300      // embed dim
#define HH   256      // hidden
#define CC_  10       // classes
#define FOURH 1024    // 4*H
#define KTE  10       // K-tiles for E padded to 320 (10 * 32)
#define KTH  8        // K-tiles for H (256 / 32)
#define NT64 64       // N-tiles over 4H (1024 / 16)
#define MT   4096     // (B*T)/16 row tiles

typedef __attribute__((ext_vector_type(16))) __bf16         v16bf;
typedef __attribute__((ext_vector_type(8)))  float          v8f;
typedef __attribute__((ext_vector_type(8)))  unsigned short v8us;
typedef __attribute__((ext_vector_type(4)))  unsigned int   v4u;
typedef __attribute__((ext_vector_type(8)))  int            v8i;
typedef __attribute__((ext_vector_type(4)))  int            v4i;

// sizes in elements (bf16 = ushort)
#define WQ_ELEMS ((size_t)KTE * NT64 * 512)          // 327,680 per direction
#define UQ_ELEMS ((size_t)KTH * NT64 * 512)          // 262,144 per direction
#define XW_ELEMS ((size_t)MT * NT64 * 256)           // 67,108,864 per direction
#define HCAT_ELEMS ((size_t)BB * TT * 512)           // 33,554,432

// Tensor Data Mover availability (device pass only; this toolchain = 6-arg form)
#if defined(__gfx1250__) && __has_builtin(__builtin_amdgcn_tensor_load_to_lds) && \
    __has_builtin(__builtin_amdgcn_s_wait_tensorcnt)
#define USE_TDM 1
#else
#define USE_TDM 0
#endif

__device__ __forceinline__ unsigned short f2bf_u(float f) {
  unsigned u = __builtin_bit_cast(unsigned, f);
  unsigned r = u + 0x7fffu + ((u >> 16) & 1u);          // RNE
  return (unsigned short)(r >> 16);
}
__device__ __forceinline__ __bf16 f2bf(float f) {
  unsigned short h = f2bf_u(f);
  return __builtin_bit_cast(__bf16, h);
}
__device__ __forceinline__ float bfu2f(unsigned short h) {
  unsigned u = ((unsigned)h) << 16;
  return __builtin_bit_cast(float, u);
}
__device__ __forceinline__ float sigm(float x) { return 1.0f / (1.0f + __expf(-x)); }

// A/B fragment K position for (element e, lane-half hi). ISA 7.12.2, 16-bit A 16x32:
// VGPR p<4: K = 2p (+8 if lane>=16); VGPR p>=4: K = 16 + 2(p-4) (+8 if lane>=16).
__device__ __forceinline__ int klocal(int e, int hi) {
  int p = e >> 1, odd = e & 1;
  int base = (p < 4) ? (p * 2) : (16 + (p - 4) * 2);
  return base + 8 * hi + odd;
}

#if USE_TDM
// One-shot 1-D TDM copy: 32 KB (4096 x 8B) contiguous global -> LDS.
// D# built per CDNA5 ISA 8.3/8.4: count=1, type=2(image), data_size=3 (8B),
// tile_dim0 = tensor_dim0 = 4096, 1-D tile (tile_dim1/2 = 0), groups 2/3 zero.
// 6-arg builtin (clang-23 / therock-10.0 headers): extra int32x8 arg zero-filled.
__device__ __forceinline__ void tdm_load_32k(const void* gsrc, unsigned lds_off) {
  unsigned long long ga = (unsigned long long)(uintptr_t)gsrc;
  v4u g0; v8i g1; v4i g2, g3; v8i g4;
  g0[0] = 1u;                                           // count=1 (valid), user mode
  g0[1] = lds_off;                                      // lds_addr (bytes)
  g0[2] = (unsigned)(ga & 0xffffffffu);                 // global_addr[31:0]
  g0[3] = (unsigned)((ga >> 32) & 0x01ffffffu) | 0x80000000u; // ga[56:32] | type=2
  const unsigned td0 = 4096u, td1 = 1u, tile0 = 4096u, st0 = 4096u, st1 = 4096u;
  g1[0] = (int)(3u << 16);                              // data_size=8B; wg_mask=0
  g1[1] = (int)((td0 & 0xffffu) << 16);                 // tensor_dim0[15:0]
  g1[2] = (int)((td0 >> 16) | ((td1 & 0xffffu) << 16)); // td0[31:16] | td1[15:0]
  g1[3] = (int)((td1 >> 16) | ((tile0 & 0xffffu) << 16)); // td1[31:16] | tile_dim0
  g1[4] = 0;                                            // tile_dim1=0, tile_dim2=0
  g1[5] = (int)st0;                                     // tensor_dim0_stride lo
  g1[6] = (int)((st1 & 0xffffu) << 16);                 // st0 hi=0 | st1[15:0]
  g1[7] = (int)(st1 >> 16);
  g2[0] = g2[1] = g2[2] = g2[3] = 0;
  g3[0] = g3[1] = g3[2] = g3[3] = 0;
  g4[0] = g4[1] = g4[2] = g4[3] = g4[4] = g4[5] = g4[6] = g4[7] = 0;
  __builtin_amdgcn_tensor_load_to_lds(g0, g1, g2, g3, g4, 0);
}
#endif

// ---------------- kernel 1: weight swizzle fp32 -> bf16 WMMA-B-fragment order ----------
__global__ void swizzle_wu_kernel(const float* __restrict__ W_fwd, const float* __restrict__ U_fwd,
                                  const float* __restrict__ W_bwd, const float* __restrict__ U_bwd,
                                  unsigned short* __restrict__ Wq, unsigned short* __restrict__ Uq) {
  size_t gid = (size_t)blockIdx.x * 256 + threadIdx.x;
  if (gid < 2 * WQ_ELEMS) {
    int dd = (int)(gid / WQ_ELEMS);
    size_t rem = gid % WQ_ELEMS;
    int frag = (int)(rem / 512);
    int lane = (int)((rem % 512) / 16);
    int e    = (int)(rem % 16);
    int kt = frag / NT64, nt = frag % NT64;
    int k = kt * 32 + klocal(e, lane >> 4);
    int n = nt * 16 + (lane & 15);
    const float* Wsrc = dd ? W_bwd : W_fwd;
    float v = (k < EE) ? Wsrc[(size_t)k * FOURH + n] : 0.0f;   // pad K 300->320
    Wq[gid] = f2bf_u(v);
  } else if (gid < 2 * WQ_ELEMS + 2 * UQ_ELEMS) {
    size_t g2 = gid - 2 * WQ_ELEMS;
    int dd = (int)(g2 / UQ_ELEMS);
    size_t rem = g2 % UQ_ELEMS;
    int frag = (int)(rem / 512);
    int lane = (int)((rem % 512) / 16);
    int e    = (int)(rem % 16);
    int kt = frag / NT64, nt = frag % NT64;
    int k = kt * 32 + klocal(e, lane >> 4);
    int n = nt * 16 + (lane & 15);
    const float* Usrc = dd ? U_bwd : U_fwd;
    Uq[g2] = f2bf_u(Usrc[(size_t)k * FOURH + n]);
  }
}

// ---------------- kernel 2: xW = emb[x] @ W  (gather fused, both directions) ------------
__global__ void gemm1_kernel(const int* __restrict__ x, const float* __restrict__ emb,
                             const unsigned short* __restrict__ Wq,
                             unsigned short* __restrict__ xW) {
  int d      = blockIdx.x >> 9;
  int mblock = blockIdx.x & 511;
  int wave   = threadIdx.x >> 5;
  int lane   = threadIdx.x & 31;
  int cc = lane & 15, hi = lane >> 4;
  int mtile = mblock * 8 + wave;               // 0..4095
  int r = mtile * 16 + cc;                     // row r = t*B + b
  int b = r & 255, t = r >> 8;
  int idx = x[b * TT + t];
  const float* erow = emb + (size_t)idx * EE;

  v16bf a[KTE];
#pragma unroll
  for (int kt = 0; kt < KTE; ++kt) {
#pragma unroll
    for (int p = 0; p < 8; ++p) {
      int k = kt * 32 + ((p < 4) ? p * 2 : 16 + (p - 4) * 2) + 8 * hi;
      float lo = (k < EE)     ? erow[k]     : 0.0f;
      float hf = (k + 1 < EE) ? erow[k + 1] : 0.0f;
      a[kt][2 * p]     = f2bf(lo);
      a[kt][2 * p + 1] = f2bf(hf);
    }
  }

  const unsigned short* WqD = Wq + (size_t)d * WQ_ELEMS;
  unsigned short*       xwD = xW + (size_t)d * XW_ELEMS;

  for (int nt = 0; nt < NT64; ++nt) {
    v8f acc;
#pragma unroll
    for (int i = 0; i < 8; ++i) acc[i] = 0.0f;
#pragma unroll
    for (int kt = 0; kt < KTE; ++kt) {
      const v16bf bfr = *(const v16bf*)(WqD + ((size_t)(kt * NT64 + nt)) * 512 + lane * 16);
      acc = __builtin_amdgcn_wmma_f32_16x16x32_bf16(false, a[kt], false, bfr,
                                                    (short)0, acc, false, false);
    }
    v8us pk;
#pragma unroll
    for (int i = 0; i < 8; ++i) pk[i] = f2bf_u(acc[i]);
    *(v8us*)(xwD + ((size_t)mtile * NT64 + nt) * 256 + lane * 8) = pk;
  }
}

// ---------------- kernel 3: persistent bidirectional LSTM recurrence -------------------
// 32 blocks x 256 thr -> (dir, batch-tile of 16). Wave w owns hidden units j in
// [32w,32w+32) across ALL FOUR gates so the gate combine stays in registers.
// Per-step xW slab (32KB contiguous) is staged into LDS by the Tensor Data Mover,
// overlapped with the activation phase; U ktiles 0..3 are pinned in VGPRs.
__global__ void birnn_kernel(const unsigned short* __restrict__ Uq,
                             const float* __restrict__ b_fwd, const float* __restrict__ b_bwd,
                             const unsigned short* __restrict__ xW,
                             unsigned short* __restrict__ Hcat) {
  __shared__ __bf16 hf[KTH * 32 * 16];          // h in A-fragment layout, 8 KB
#if USE_TDM
  __shared__ unsigned short xwbuf[NT64 * 256];  // one time-step xW slab, 32 KB
#endif

  int d     = blockIdx.x >> 4;
  int btile = blockIdx.x & 15;
  int w     = threadIdx.x >> 5;
  int lane  = threadIdx.x & 31;
  int cc = lane & 15, hi = lane >> 4;

  const float* bias = d ? b_bwd : b_fwd;
  int   ntl[8];                                 // order: i0 i1 f0 f1 g0 g1 o0 o1
  float bv[8];
#pragma unroll
  for (int g = 0; g < 4; ++g)
#pragma unroll
    for (int t2 = 0; t2 < 2; ++t2) {
      int nt = g * 16 + 2 * w + t2;
      ntl[g * 2 + t2] = nt;
      bv[g * 2 + t2]  = bias[nt * 16 + cc];
    }

  const unsigned short* UqD = Uq + (size_t)d * UQ_ELEMS;
  const unsigned short* xwD = xW + (size_t)d * XW_ELEMS;

  // Pin U ktiles 0..3 (32 fragments, 256 VGPRs) for the whole sequence.
  v16bf ureg[4][8];
#pragma unroll
  for (int kt = 0; kt < 4; ++kt)
#pragma unroll
    for (int i = 0; i < 8; ++i)
      ureg[kt][i] = *(const v16bf*)(UqD + ((size_t)(kt * NT64 + ntl[i])) * 512 + lane * 16);

  float cst[16];
#pragma unroll
  for (int i = 0; i < 16; ++i) cst[i] = 0.0f;
  for (int i = threadIdx.x; i < KTH * 32 * 16; i += 256) hf[i] = f2bf(0.0f);

#if USE_TDM
  unsigned xw_lds = (unsigned)(uintptr_t)(void*)xwbuf;
  if (w == 0) {   // stage step 0 (wave-uniform; TDM ignores EXEC)
    int t0 = d ? (TT - 1) : 0;
    tdm_load_32k(xwD + ((size_t)(t0 * 16 + btile) * NT64) * 256, xw_lds);
    __builtin_amdgcn_s_wait_tensorcnt((short)0);
  }
#endif
  __syncthreads();

  for (int s = 0; s < TT; ++s) {
    int tt = d ? (TT - 1 - s) : s;
    int mtile = tt * 16 + btile;

    // A fragments (shared h) from LDS
    v16bf a[KTH];
#pragma unroll
    for (int kt = 0; kt < KTH; ++kt)
      a[kt] = *(const v16bf*)&hf[(kt * 32 + lane) * 16];

    // acc = xW(t) + bias
    v8f acc[8];
#pragma unroll
    for (int i = 0; i < 8; ++i) {
#if USE_TDM
      v8us xv = *(const v8us*)&xwbuf[ntl[i] * 256 + lane * 8];
#else
      v8us xv = *(const v8us*)(xwD + ((size_t)mtile * NT64 + ntl[i]) * 256 + lane * 8);
#endif
#pragma unroll
      for (int r2 = 0; r2 < 8; ++r2) acc[i][r2] = bv[i] + bfu2f(xv[r2]);
    }
#if !USE_TDM
    if (s + 1 < TT) {
      int tn = d ? (TT - 2 - s) : (s + 1);
      __builtin_prefetch(xwD + ((size_t)(tn * 16 + btile) * NT64 + ntl[0]) * 256, 0, 1);
    }
#endif
    (void)mtile;

    // z += h @ U : ktiles 0..3 from registers, 4..7 streamed from L2
#pragma unroll
    for (int kt = 0; kt < 4; ++kt)
#pragma unroll
      for (int i = 0; i < 8; ++i)
        acc[i] = __builtin_amdgcn_wmma_f32_16x16x32_bf16(false, a[kt], false, ureg[kt][i],
                                                         (short)0, acc[i], false, false);
#pragma unroll
    for (int kt = 4; kt < KTH; ++kt)
#pragma unroll
      for (int i = 0; i < 8; ++i) {
        const v16bf bfr = *(const v16bf*)(UqD + ((size_t)(kt * NT64 + ntl[i])) * 512 + lane * 16);
        acc[i] = __builtin_amdgcn_wmma_f32_16x16x32_bf16(false, a[kt], false, bfr,
                                                         (short)0, acc[i], false, false);
      }
    __syncthreads();          // all reads of hf (and xwbuf) complete

#if USE_TDM
    if (w == 0 && s + 1 < TT) {   // overlap next-step TDM with activation phase
      int tn = d ? (TT - 2 - s) : (s + 1);
      tdm_load_32k(xwD + ((size_t)(tn * 16 + btile) * NT64) * 256, xw_lds);
    }
#endif

    // gates in registers; write new h to LDS (A-frag layout) and Hcat (bf16)
#pragma unroll
    for (int t2 = 0; t2 < 2; ++t2) {
      int j = 32 * w + t2 * 16 + cc;
      int lane_hi = 16 * ((cc >> 3) & 1);
      int e = (t2 * 4 + ((cc & 7) >> 1)) * 2 + (cc & 1);
#pragma unroll
      for (int r2 = 0; r2 < 8; ++r2) {
        int m  = r2 + 8 * hi;
        float iv = sigm(acc[0 + t2][r2]);
        float fv = sigm(acc[2 + t2][r2]);
        float gv = fmaxf(acc[4 + t2][r2], 0.0f);
        float ov = sigm(acc[6 + t2][r2]);
        int ci = t2 * 8 + r2;
        cst[ci] = fv * cst[ci] + iv * gv;
        float hv = ov * fmaxf(cst[ci], 0.0f);
        hf[(w * 32 + (m + lane_hi)) * 16 + e] = f2bf(hv);
        int bg = btile * 16 + m;
        Hcat[((size_t)bg * TT + tt) * 512 + d * HH + j] = f2bf_u(hv);
      }
    }

#if USE_TDM
    if (w == 0 && s + 1 < TT) __builtin_amdgcn_s_wait_tensorcnt((short)0);
#endif
    __syncthreads();
  }
}

// ---------------- kernel 4: output layer + softmax --------------------------------------
__global__ void out_kernel(const unsigned short* __restrict__ Hcat,
                           const float* __restrict__ W_out, const float* __restrict__ b_out,
                           float* __restrict__ out) {
  __shared__ float red[256];
  __shared__ float tot[CC_];
  int b = blockIdx.x, tid = threadIdx.x;
  float acc[CC_];
#pragma unroll
  for (int c = 0; c < CC_; ++c) acc[c] = 0.0f;
  const unsigned short* hrow = Hcat + (size_t)b * (TT * 512);
  for (int k = tid; k < TT * 512; k += 256) {
    float hv = bfu2f(hrow[k]);
    const float* wr = W_out + (size_t)k * CC_;
#pragma unroll
    for (int c = 0; c < CC_; ++c) acc[c] = fmaf(hv, wr[c], acc[c]);
  }
  for (int c = 0; c < CC_; ++c) {
    red[tid] = acc[c];
    __syncthreads();
    for (int off = 128; off > 0; off >>= 1) {
      if (tid < off) red[tid] += red[tid + off];
      __syncthreads();
    }
    if (tid == 0) tot[c] = red[0] + b_out[c];
    __syncthreads();
  }
  if (tid == 0) {
    float mx = tot[0];
#pragma unroll
    for (int c = 1; c < CC_; ++c) mx = fmaxf(mx, tot[c]);
    float ex[CC_], sum = 0.0f;
#pragma unroll
    for (int c = 0; c < CC_; ++c) { ex[c] = __expf(tot[c] - mx); sum += ex[c]; }
#pragma unroll
    for (int c = 0; c < CC_; ++c) out[b * CC_ + c] = ex[c] / sum;
  }
}

// ---------------- launcher --------------------------------------------------------------
extern "C" void kernel_launch(void* const* d_in, const int* in_sizes, int n_in,
                              void* d_out, int out_size, void* d_ws, size_t ws_size,
                              hipStream_t stream) {
  const int*   x     = (const int*)d_in[0];
  const float* emb   = (const float*)d_in[1];
  const float* W_fwd = (const float*)d_in[2];
  const float* U_fwd = (const float*)d_in[3];
  const float* b_fwd = (const float*)d_in[4];
  const float* W_bwd = (const float*)d_in[5];
  const float* U_bwd = (const float*)d_in[6];
  const float* b_bwd = (const float*)d_in[7];
  const float* W_out = (const float*)d_in[8];
  const float* b_out = (const float*)d_in[9];
  float* out = (float*)d_out;

  char* ws = (char*)d_ws;
  unsigned short* Wq   = (unsigned short*)ws;                 ws += 2 * WQ_ELEMS * 2;
  unsigned short* Uq   = (unsigned short*)ws;                 ws += 2 * UQ_ELEMS * 2;
  unsigned short* xW   = (unsigned short*)ws;                 ws += 2 * XW_ELEMS * 2;
  unsigned short* Hcat = (unsigned short*)ws;                 // + HCAT_ELEMS*2

  size_t swz_threads = 2 * WQ_ELEMS + 2 * UQ_ELEMS;           // 1,179,648
  int swz_blocks = (int)((swz_threads + 255) / 256);
  swizzle_wu_kernel<<<swz_blocks, 256, 0, stream>>>(W_fwd, U_fwd, W_bwd, U_bwd, Wq, Uq);
  gemm1_kernel<<<1024, 256, 0, stream>>>(x, emb, Wq, xW);
  birnn_kernel<<<32, 256, 0, stream>>>(Uq, b_fwd, b_bwd, xW, Hcat);
  out_kernel<<<BB, 256, 0, stream>>>(Hcat, W_out, b_out, out);
}
